// CustomSparseGLU_7327214207122
// MI455X (gfx1250) — matrix operand order repaired and checked by reading, not compiled, not run on hardware
//
#include <hip/hip_runtime.h>
#include <hip/hip_bf16.h>
#include <math.h>

#define T_TOK 16384
#define HID   1024
#define FFN   2048
#define NEXP  8
#define CAP   (T_TOK / NEXP)   // 2048 tokens per expert

typedef __attribute__((ext_vector_type(16))) __bf16 v16bf;
typedef __attribute__((ext_vector_type(8)))  __bf16 v8bf;
typedef __attribute__((ext_vector_type(4)))  __bf16 v4bf;
typedef __attribute__((ext_vector_type(8)))  float  v8f;
typedef __attribute__((ext_vector_type(4)))  float  v4f;

// ---- fragment builders -----------------------------------------------------
// A-frag 16x32 bf16 (wave32): lane holds row (lane&15); elems 0..15 = K[kb..kb+15],
// kb = k0 + (lane>=16 ? 16 : 0).
// B-frag 32x16 bf16: lane holds col (lane&15); elems 0..15 = K[kb..kb+15],
// kb = k0 + (lane>=16 ? 16 : 0).  All map to contiguous memory runs here.

__device__ __forceinline__ v16bf frag_from_f32_pair(const float* c0p, const float* c1p) {
  v8f c0 = *(const v8f*)c0p;
  v8f c1 = *(const v8f*)c1p;
  v16bf r;
#pragma unroll
  for (int i = 0; i < 8; ++i) { r[i] = (__bf16)c0[i]; r[i + 8] = (__bf16)c1[i]; }
  return r;
}

__device__ __forceinline__ v16bf frag_from_bf16_pair(const __bf16* c0p, const __bf16* c1p) {
  v8bf c0 = *(const v8bf*)c0p;
  v8bf c1 = *(const v8bf*)c1p;
  v16bf r;
#pragma unroll
  for (int i = 0; i < 8; ++i) { r[i] = c0[i]; r[i + 8] = c1[i]; }
  return r;
}

__device__ __forceinline__ float gelu_exact(float x) {
  return 0.5f * x * (1.0f + erff(x * 0.70710678118654752f));
}

// ---- kernel 0: elementwise f32 -> bf16 pack (one sweep; keeps cvt out of GEMMs)
__global__ void __launch_bounds__(256)
pack_f32_to_bf16(const float* __restrict__ src, __bf16* __restrict__ dst, int n4) {
  const int i = blockIdx.x * 256 + threadIdx.x;
  if (i < n4) {
    const v4f f = ((const v4f*)src)[i];
    v4bf o;
#pragma unroll
    for (int j = 0; j < 4; ++j) o[j] = (__bf16)f[j];
    ((v4bf*)dst)[i] = o;
  }
}

// ---- kernel 1: w2 [E,F,H] f32 -> w2t [E,H,F] bf16 (LDS-tiled transpose) ----
__global__ void __launch_bounds__(256)
w2_transpose_bf16(const float* __restrict__ w2, __bf16* __restrict__ w2t) {
  __shared__ float tile[32][33];
  const int e  = blockIdx.z;
  const int h0 = blockIdx.x * 32;
  const int f0 = blockIdx.y * 32;
  const int tx = threadIdx.x;   // 0..31
  const int ty = threadIdx.y;   // 0..7
  const float* src = w2 + ((size_t)e * FFN + f0) * HID + h0;
#pragma unroll
  for (int j = 0; j < 4; ++j)
    tile[ty + 8 * j][tx] = src[(size_t)(ty + 8 * j) * HID + tx];
  __syncthreads();
  __bf16* dst = w2t + ((size_t)e * HID + h0) * FFN + f0;
#pragma unroll
  for (int j = 0; j < 4; ++j)
    dst[(size_t)(ty + 8 * j) * FFN + tx] = (__bf16)tile[tx][ty + 8 * j];
}

// ---- kernel 2a: fused SDD dual-GEMM + GLU, bf16 operands (main path) ------
// Per expert: x1 = x.w1^T, x2 = x.v1^T over K=HID, h = gelu(x1)*x2 -> bf16.
// Block tile 128x64 (4 waves, 2x2), wave tile 64x32 (4Mx2N, dual gates:
// 16 WMMA per K-step vs 8 fragment loads).
__global__ void __launch_bounds__(128)
glu_sdd_wmma_bf16(const __bf16* __restrict__ xb, const __bf16* __restrict__ w1b,
                  const __bf16* __restrict__ v1b, __bf16* __restrict__ hbuf) {
  const int e     = blockIdx.z;
  const int row0  = blockIdx.y * 128;  // token tile within expert
  const int col0  = blockIdx.x * 64;   // ffn tile
  const int wave  = threadIdx.x >> 5;
  const int lane  = threadIdx.x & 31;
  const int waveM = (wave >> 1) * 64;
  const int waveN = (wave & 1) * 32;
  const int lhalf = lane >> 4;
  const int l16   = lane & 15;

  const __bf16* arow[4];
#pragma unroll
  for (int mi = 0; mi < 4; ++mi) {
    const int tok = e * CAP + row0 + waveM + mi * 16 + l16;
    arow[mi] = xb + (size_t)tok * HID;
  }
  const __bf16* b1row[2];
  const __bf16* b2row[2];
#pragma unroll
  for (int ni = 0; ni < 2; ++ni) {
    const int f = col0 + waveN + ni * 16 + l16;
    const size_t off = ((size_t)e * FFN + f) * HID;
    b1row[ni] = w1b + off;
    b2row[ni] = v1b + off;
  }

  v8f acc1[4][2], acc2[4][2];
#pragma unroll
  for (int mi = 0; mi < 4; ++mi)
#pragma unroll
    for (int ni = 0; ni < 2; ++ni)
#pragma unroll
      for (int i = 0; i < 8; ++i) { acc1[mi][ni][i] = 0.f; acc2[mi][ni][i] = 0.f; }

#pragma unroll 2
  for (int k0 = 0; k0 < HID; k0 += 32) {
    const int akb = k0 + lhalf * 16;
    const int bkb = k0 + lhalf * 16;
    v16bf a[4], b1[2], b2[2];
#pragma unroll
    for (int mi = 0; mi < 4; ++mi)
      a[mi] = frag_from_bf16_pair(arow[mi] + akb, arow[mi] + akb + 8);
#pragma unroll
    for (int ni = 0; ni < 2; ++ni) {
      b1[ni] = *(const v16bf*)(b1row[ni] + bkb);
      b2[ni] = *(const v16bf*)(b2row[ni] + bkb);
    }
#pragma unroll
    for (int mi = 0; mi < 4; ++mi)
#pragma unroll
      for (int ni = 0; ni < 2; ++ni) {
        acc1[mi][ni] = __builtin_amdgcn_wmma_f32_16x16x32_bf16(
            false, a[mi], false, b1[ni], (short)0, acc1[mi][ni], false, false);
        acc2[mi][ni] = __builtin_amdgcn_wmma_f32_16x16x32_bf16(
            false, a[mi], false, b2[ni], (short)0, acc2[mi][ni], false, false);
      }
  }

  // GLU epilogue: C/D layout -> VGPR r, lanes 0-15: M=r, lanes 16-31: M=r+8
#pragma unroll
  for (int mi = 0; mi < 4; ++mi)
#pragma unroll
    for (int ni = 0; ni < 2; ++ni) {
      const int colg = col0 + waveN + ni * 16 + l16;
#pragma unroll
      for (int r = 0; r < 8; ++r) {
        const int m   = waveM + mi * 16 + r + lhalf * 8;
        const int tok = e * CAP + row0 + m;
        const float g = gelu_exact(acc1[mi][ni][r]) * acc2[mi][ni][r];
        hbuf[(size_t)tok * FFN + colg] = (__bf16)g;
      }
    }
}

// ---- kernel 2b: fallback SDD dual-GEMM + GLU with in-register f32->bf16 ----
// Used only when workspace is too small for the packed operand copies.
__global__ void __launch_bounds__(128)
glu_sdd_wmma_f32(const float* __restrict__ x, const float* __restrict__ w1,
                 const float* __restrict__ v1, __bf16* __restrict__ hbuf) {
  const int e     = blockIdx.z;
  const int row0  = blockIdx.y * 64;
  const int col0  = blockIdx.x * 64;
  const int wave  = threadIdx.x >> 5;
  const int lane  = threadIdx.x & 31;
  const int waveM = (wave >> 1) * 32;
  const int waveN = (wave & 1) * 32;
  const int lhalf = lane >> 4;
  const int l16   = lane & 15;

  const float* arow[2];
#pragma unroll
  for (int mi = 0; mi < 2; ++mi) {
    const int tok = e * CAP + row0 + waveM + mi * 16 + l16;
    arow[mi] = x + (size_t)tok * HID;
  }
  const float* b1row[2];
  const float* b2row[2];
#pragma unroll
  for (int ni = 0; ni < 2; ++ni) {
    const int f = col0 + waveN + ni * 16 + l16;
    const size_t off = ((size_t)e * FFN + f) * HID;
    b1row[ni] = w1 + off;
    b2row[ni] = v1 + off;
  }

  v8f acc1[2][2], acc2[2][2];
#pragma unroll
  for (int mi = 0; mi < 2; ++mi)
#pragma unroll
    for (int ni = 0; ni < 2; ++ni)
#pragma unroll
      for (int i = 0; i < 8; ++i) { acc1[mi][ni][i] = 0.f; acc2[mi][ni][i] = 0.f; }

#pragma unroll 2
  for (int k0 = 0; k0 < HID; k0 += 32) {
    const int akb = k0 + lhalf * 16;
    const int bkb = k0 + lhalf * 16;
    v16bf a[2], b1[2], b2[2];
#pragma unroll
    for (int mi = 0; mi < 2; ++mi)
      a[mi] = frag_from_f32_pair(arow[mi] + akb, arow[mi] + akb + 8);
#pragma unroll
    for (int ni = 0; ni < 2; ++ni) {
      b1[ni] = frag_from_f32_pair(b1row[ni] + bkb, b1row[ni] + bkb + 8);
      b2[ni] = frag_from_f32_pair(b2row[ni] + bkb, b2row[ni] + bkb + 8);
    }
#pragma unroll
    for (int mi = 0; mi < 2; ++mi)
#pragma unroll
      for (int ni = 0; ni < 2; ++ni) {
        acc1[mi][ni] = __builtin_amdgcn_wmma_f32_16x16x32_bf16(
            false, a[mi], false, b1[ni], (short)0, acc1[mi][ni], false, false);
        acc2[mi][ni] = __builtin_amdgcn_wmma_f32_16x16x32_bf16(
            false, a[mi], false, b2[ni], (short)0, acc2[mi][ni], false, false);
      }
  }

#pragma unroll
  for (int mi = 0; mi < 2; ++mi)
#pragma unroll
    for (int ni = 0; ni < 2; ++ni) {
      const int colg = col0 + waveN + ni * 16 + l16;
#pragma unroll
      for (int r = 0; r < 8; ++r) {
        const int m   = waveM + mi * 16 + r + lhalf * 8;
        const int tok = e * CAP + row0 + m;
        const float g = gelu_exact(acc1[mi][ni][r]) * acc2[mi][ni][r];
        hbuf[(size_t)tok * FFN + colg] = (__bf16)g;
      }
    }
}

// ---- kernel 3: DSD GEMM out = h . w2 over K=FFN ---------------------------
// Block tile 128x64 (4 waves, 2x2), wave tile 64x32 (4x2 WMMA tiles).
__global__ void __launch_bounds__(128)
dsd_wmma(const __bf16* __restrict__ hbuf, const __bf16* __restrict__ w2t,
         float* __restrict__ out) {
  const int e     = blockIdx.z;
  const int row0  = blockIdx.y * 128;
  const int col0  = blockIdx.x * 64;
  const int wave  = threadIdx.x >> 5;
  const int lane  = threadIdx.x & 31;
  const int waveM = (wave >> 1) * 64;
  const int waveN = (wave & 1) * 32;
  const int lhalf = lane >> 4;
  const int l16   = lane & 15;

  const __bf16* arow[4];
#pragma unroll
  for (int mi = 0; mi < 4; ++mi) {
    const int tok = e * CAP + row0 + waveM + mi * 16 + l16;
    arow[mi] = hbuf + (size_t)tok * FFN;
  }
  const __bf16* brow[2];
#pragma unroll
  for (int ni = 0; ni < 2; ++ni) {
    const int hh = col0 + waveN + ni * 16 + l16;
    brow[ni] = w2t + ((size_t)e * HID + hh) * FFN;
  }

  v8f acc[4][2];
#pragma unroll
  for (int mi = 0; mi < 4; ++mi)
#pragma unroll
    for (int ni = 0; ni < 2; ++ni)
#pragma unroll
      for (int i = 0; i < 8; ++i) acc[mi][ni][i] = 0.f;

#pragma unroll 2
  for (int k0 = 0; k0 < FFN; k0 += 32) {
    const int akb = k0 + lhalf * 16;
    const int bkb = k0 + lhalf * 16;
    v16bf a[4], b[2];
#pragma unroll
    for (int mi = 0; mi < 4; ++mi)
      a[mi] = frag_from_bf16_pair(arow[mi] + akb, arow[mi] + akb + 8);
#pragma unroll
    for (int ni = 0; ni < 2; ++ni)
      b[ni] = *(const v16bf*)(brow[ni] + bkb);   // 32B aligned contiguous run
#pragma unroll
    for (int mi = 0; mi < 4; ++mi)
#pragma unroll
      for (int ni = 0; ni < 2; ++ni)
        acc[mi][ni] = __builtin_amdgcn_wmma_f32_16x16x32_bf16(
            false, a[mi], false, b[ni], (short)0, acc[mi][ni], false, false);
  }

#pragma unroll
  for (int mi = 0; mi < 4; ++mi)
#pragma unroll
    for (int ni = 0; ni < 2; ++ni) {
      const int colg = col0 + waveN + ni * 16 + l16;
#pragma unroll
      for (int r = 0; r < 8; ++r) {
        const int m   = waveM + mi * 16 + r + lhalf * 8;
        const int tok = e * CAP + row0 + m;
        out[(size_t)tok * HID + colg] = acc[mi][ni][r];
      }
    }
}

// ---- launcher --------------------------------------------------------------
extern "C" void kernel_launch(void* const* d_in, const int* in_sizes, int n_in,
                              void* d_out, int out_size, void* d_ws, size_t ws_size,
                              hipStream_t stream) {
  (void)in_sizes; (void)n_in; (void)out_size;
  const float* x  = (const float*)d_in[0];
  const float* w1 = (const float*)d_in[1];
  const float* v1 = (const float*)d_in[2];
  const float* w2 = (const float*)d_in[3];
  // d_in[4] (expert_ids) unused: tokens are pre-sorted with equal capacity.
  float* out = (float*)d_out;

  const size_t SZ_W2T = (size_t)NEXP * HID * FFN * sizeof(__bf16);   // 32 MiB
  const size_t SZ_H   = (size_t)T_TOK * FFN * sizeof(__bf16);        // 64 MiB
  const size_t SZ_X   = (size_t)T_TOK * HID * sizeof(__bf16);        // 32 MiB
  const size_t SZ_W   = (size_t)NEXP * FFN * HID * sizeof(__bf16);   // 32 MiB

  char* p = (char*)d_ws;
  __bf16* w2t  = (__bf16*)(p);
  __bf16* hbuf = (__bf16*)(p + SZ_W2T);

  w2_transpose_bf16<<<dim3(HID / 32, FFN / 32, NEXP), dim3(32, 8), 0, stream>>>(w2, w2t);

  const bool packed = ws_size >= SZ_W2T + SZ_H + SZ_X + 2 * SZ_W;    // 192 MiB
  if (packed) {
    __bf16* xb  = (__bf16*)(p + SZ_W2T + SZ_H);
    __bf16* w1b = (__bf16*)(p + SZ_W2T + SZ_H + SZ_X);
    __bf16* v1b = (__bf16*)(p + SZ_W2T + SZ_H + SZ_X + SZ_W);
    const int nx4 = (int)(T_TOK * (size_t)HID / 4);
    const int nw4 = (int)((size_t)NEXP * FFN * HID / 4);
    pack_f32_to_bf16<<<(nx4 + 255) / 256, 256, 0, stream>>>(x,  xb,  nx4);
    pack_f32_to_bf16<<<(nw4 + 255) / 256, 256, 0, stream>>>(w1, w1b, nw4);
    pack_f32_to_bf16<<<(nw4 + 255) / 256, 256, 0, stream>>>(v1, v1b, nw4);
    glu_sdd_wmma_bf16<<<dim3(FFN / 64, CAP / 128, NEXP), 128, 0, stream>>>(xb, w1b, v1b, hbuf);
  } else {
    glu_sdd_wmma_f32<<<dim3(FFN / 64, CAP / 64, NEXP), 128, 0, stream>>>(x, w1, v1, hbuf);
  }

  dsd_wmma<<<dim3(HID / 64, CAP / 128, NEXP), 128, 0, stream>>>(hbuf, w2t, out);
}
